// MixedOp_18425409699935
// MI455X (gfx1250) — compile-verified, matrix operand order checked by reading, not run on hardware
//
#include <hip/hip_runtime.h>
#include <hip/hip_bf16.h>

typedef __attribute__((ext_vector_type(16))) _Float16 v16h;
typedef __attribute__((ext_vector_type(8)))  float    v8f;

#define TPB 256

// ---------------------------------------------------------------- reductions
__device__ __forceinline__ float blk_sum(float v, float* red) {
  int t = threadIdx.x;
  red[t] = v; __syncthreads();
  for (int s = TPB / 2; s > 0; s >>= 1) {
    if (t < s) red[t] += red[t + s];
    __syncthreads();
  }
  float r = red[0]; __syncthreads();
  return r;
}

__device__ __forceinline__ float blk_max(float v, float* red) {
  int t = threadIdx.x;
  red[t] = v; __syncthreads();
  for (int s = TPB / 2; s > 0; s >>= 1) {
    if (t < s) red[t] = fmaxf(red[t], red[t + s]);
    __syncthreads();
  }
  float r = red[0]; __syncthreads();
  return r;
}

// ---------------------------------------------------------------- WMMA GEMM
// C[b][m][n] = sum_k A[b][m][k] * Bm[b][n][k]  (both operands row-major along K)
// Each wave computes a 16x64 strip: one A fragment reused across 4 B tiles,
// 4 independent accumulators -> 4 pipelined WMMAs per k-step.
// Requires M%16==0, N%64==0, K%32==0.
__global__ void __launch_bounds__(TPB)
wmma_gemm(const _Float16* __restrict__ A, long long sA, int lda,
          const _Float16* __restrict__ Bm, long long sB, int ldb,
          const float* __restrict__ bias,
          float* __restrict__ Cf, long long sCf, int ldc, int accum,
          _Float16* __restrict__ Ch, long long sCh, int ldch, int transC,
          int M, int N, int K, int batch)
{
  int wave = blockIdx.x * (TPB >> 5) + (threadIdx.x >> 5);
  int Mt = M >> 4, Ng = N >> 6;
  int tiles = batch * Mt * Ng;
  if (wave >= tiles) return;
  int b  = wave / (Mt * Ng);
  int tt = wave % (Mt * Ng);
  int m0 = (tt / Ng) << 4;
  int n0 = (tt % Ng) << 6;
  int lane = threadIdx.x & 31;
  int lm = lane & 15, lh = lane >> 4;

  const _Float16* ap  = A  + (size_t)b * sA + (size_t)(m0 + lm) * lda + lh * 16;
  const _Float16* bp0 = Bm + (size_t)b * sB + (size_t)(n0 + lm) * ldb + lh * 16;
  const _Float16* bp1 = bp0 + (size_t)16 * ldb;
  const _Float16* bp2 = bp0 + (size_t)32 * ldb;
  const _Float16* bp3 = bp0 + (size_t)48 * ldb;

  v8f acc[4];
  #pragma unroll
  for (int g = 0; g < 4; g++)
    #pragma unroll
    for (int r = 0; r < 8; r++) acc[g][r] = 0.0f;

  if (Cf && accum) {
    #pragma unroll
    for (int g = 0; g < 4; g++)
      #pragma unroll
      for (int r = 0; r < 8; r++)
        acc[g][r] = Cf[(size_t)b * sCf + (size_t)(m0 + r + 8 * lh) * ldc
                       + n0 + g * 16 + lm];
  }

  for (int k0 = 0; k0 < K; k0 += 32) {
    v16h av  = *(const v16h*)(ap  + k0);
    v16h b0v = *(const v16h*)(bp0 + k0);
    v16h b1v = *(const v16h*)(bp1 + k0);
    v16h b2v = *(const v16h*)(bp2 + k0);
    v16h b3v = *(const v16h*)(bp3 + k0);
    acc[0] = __builtin_amdgcn_wmma_f32_16x16x32_f16(false, av, false, b0v,
                                                    (short)0, acc[0], false, false);
    acc[1] = __builtin_amdgcn_wmma_f32_16x16x32_f16(false, av, false, b1v,
                                                    (short)0, acc[1], false, false);
    acc[2] = __builtin_amdgcn_wmma_f32_16x16x32_f16(false, av, false, b2v,
                                                    (short)0, acc[2], false, false);
    acc[3] = __builtin_amdgcn_wmma_f32_16x16x32_f16(false, av, false, b3v,
                                                    (short)0, acc[3], false, false);
  }

  #pragma unroll
  for (int g = 0; g < 4; g++) {
    int nn = n0 + g * 16 + lm;
    float bb = bias ? bias[nn] : 0.0f;
    #pragma unroll
    for (int r = 0; r < 8; r++) {
      int mm = m0 + r + 8 * lh;
      float v = acc[g][r] + bb;
      if (Cf) Cf[(size_t)b * sCf + (size_t)mm * ldc + nn] = v;
      if (Ch) {
        _Float16 hv = (_Float16)v;
        if (transC) Ch[(size_t)b * sCh + (size_t)nn * ldch + mm] = hv;
        else        Ch[(size_t)b * sCh + (size_t)mm * ldch + nn] = hv;
      }
    }
  }
}

// ---------------------------------------------------------------- diag prep
// dd[c] = diag_of(adj)[c]   (dense, so the scale pass reads coalesced)
__global__ void pack_diag(const float* __restrict__ adj, int dstride,
                          float* __restrict__ dd, int n)
{
  int i = blockIdx.x * TPB + threadIdx.x;
  if (i < n) dd[i] = adj[(size_t)i * dstride];
}

// dst[i] = src[i] * dd[i % Crow]
__global__ void diag_scale(const _Float16* __restrict__ src,
                           const float* __restrict__ dd,
                           _Float16* __restrict__ dst, int Crow, size_t total)
{
  size_t i = blockIdx.x * (size_t)TPB + threadIdx.x;
  if (i >= total) return;
  int c = (int)(i % Crow);
  dst[i] = (_Float16)((float)src[i] * dd[c]);
}

// ---------------------------------------------------------------- row prep
__global__ void __launch_bounds__(TPB)
rownorm_prep(const float* __restrict__ X, int Crow,
             int rowsSrcPerBatch, int rowsDstPerBatch,
             _Float16* __restrict__ nfH, _Float16* __restrict__ rawH,
             _Float16* __restrict__ rawT, int ldT, int totalRows)
{
  __shared__ float red[TPB];
  int row = blockIdx.x;
  if (row >= totalRows) return;
  int b = row / rowsSrcPerBatch, j = row % rowsSrcPerBatch;
  const float* x = X + (size_t)row * Crow;
  float ss = 0;
  for (int c = threadIdx.x; c < Crow; c += TPB) { float v = x[c]; ss += v * v; }
  ss = blk_sum(ss, red);
  float inv = 1.0f / fmaxf(sqrtf(ss), 1e-12f);
  size_t drow = (size_t)b * rowsDstPerBatch + j;
  for (int c = threadIdx.x; c < Crow; c += TPB) {
    float v = x[c];
    if (nfH)  nfH[drow * Crow + c] = (_Float16)(v * inv);
    if (rawH) rawH[drow * Crow + c] = (_Float16)v;
    if (rawT) rawT[(size_t)b * Crow * ldT + (size_t)c * ldT + j] = (_Float16)v;
  }
}

// out = softmax(5*A[0:validN]) per row; cols >= validN written as 0 (f16)
__global__ void __launch_bounds__(TPB)
softmax5(const float* __restrict__ A, int ld, int validN, _Float16* __restrict__ outH)
{
  __shared__ float red[TPB];
  int row = blockIdx.x;
  const float* a = A + (size_t)row * ld;
  float m = -3.4e38f;
  for (int c = threadIdx.x; c < validN; c += TPB) m = fmaxf(m, a[c]);
  m = blk_max(m, red);
  float s = 0;
  for (int c = threadIdx.x; c < validN; c += TPB) s += expf(5.0f * (a[c] - m));
  s = blk_sum(s, red);
  float inv = 1.0f / s;
  for (int c = threadIdx.x; c < ld; c += TPB) {
    float e = (c < validN) ? expf(5.0f * (a[c] - m)) * inv : 0.0f;
    outH[(size_t)row * ld + c] = (_Float16)e;
  }
}

// diff_prop row op: exp(5(a-max)), zero diagonal, L1-normalize, emit s=rowsum
__global__ void __launch_bounds__(TPB)
dpnorm(const float* __restrict__ A, int N, _Float16* __restrict__ outH,
       float* __restrict__ srow)
{
  __shared__ float red[TPB];
  int row = blockIdx.x;
  int j = row % N;
  const float* a = A + (size_t)row * N;
  float m = -3.4e38f;
  for (int c = threadIdx.x; c < N; c += TPB) m = fmaxf(m, a[c]);
  m = blk_max(m, red);
  float s = 0;
  for (int c = threadIdx.x; c < N; c += TPB)
    s += (c == j) ? 0.0f : expf(5.0f * (a[c] - m));
  s = blk_sum(s, red);
  float d = fmaxf(s, 1e-12f);
  float inv = 1.0f / d;
  if (threadIdx.x == 0) srow[row] = s * inv;
  for (int c = threadIdx.x; c < N; c += TPB) {
    float e = (c == j) ? 0.0f : expf(5.0f * (a[c] - m)) * inv;
    outH[(size_t)row * N + c] = (_Float16)e;
  }
}

// layernorm + leaky_relu; value = srow ? s*in0 - in1 : in0
__global__ void __launch_bounds__(TPB)
ln_lrelu(const float* __restrict__ in0, const float* __restrict__ in1,
         const float* __restrict__ srow, const float* __restrict__ g,
         const float* __restrict__ be, float* __restrict__ out, int Crow)
{
  __shared__ float red[TPB];
  int row = blockIdx.x;
  const float* a0 = in0 + (size_t)row * Crow;
  const float* a1 = in1 ? in1 + (size_t)row * Crow : nullptr;
  float sc = srow ? srow[row] : 1.0f;
  float s = 0, s2 = 0;
  for (int c = threadIdx.x; c < Crow; c += TPB) {
    float v = sc * a0[c] - (a1 ? a1[c] : 0.0f);
    s += v; s2 += v * v;
  }
  s  = blk_sum(s, red);
  s2 = blk_sum(s2, red);
  float m = s / Crow;
  float var = s2 / Crow - m * m;
  float inv = rsqrtf(var + 1e-5f);
  for (int c = threadIdx.x; c < Crow; c += TPB) {
    float v = sc * a0[c] - (a1 ? a1[c] : 0.0f);
    float o = (v - m) * inv * g[c] + be[c];
    out[(size_t)row * Crow + c] = o > 0.0f ? o : 0.01f * o;
  }
}

// node_att factored: v[b][k] = sum_m attw[m] * nf[b][m][k]
__global__ void natt_v(const _Float16* __restrict__ nf, const float* __restrict__ attw,
                       float* __restrict__ v, int NT, int Crow, int total)
{
  int i = blockIdx.x * TPB + threadIdx.x;
  if (i >= total) return;
  int b = i / Crow, k = i % Crow;
  const _Float16* base = nf + (size_t)b * NT * Crow + k;
  float acc = 0;
  for (int m = 0; m < NT; m++) acc += attw[m] * (float)base[(size_t)m * Crow];
  v[i] = acc;
}

// att = sigmoid(nf[row].v[b] + pos.attw_pos + b); out += wvec[widx]*fain*att
__global__ void __launch_bounds__(TPB)
natt_apply(const _Float16* __restrict__ nf, const float* __restrict__ v,
           const float* __restrict__ pos, int posD,
           const float* __restrict__ attw, const float* __restrict__ attb,
           const float* __restrict__ fain, const float* __restrict__ wvec, int widx,
           float* __restrict__ out, int Crow, int NT)
{
  __shared__ float red[TPB];
  __shared__ float satt;
  int row = blockIdx.x;
  int b = row / NT;
  const _Float16* nr = nf + (size_t)row * Crow;
  const float* vb = v + (size_t)b * Crow;
  float acc = 0;
  for (int c = threadIdx.x; c < Crow; c += TPB) acc += (float)nr[c] * vb[c];
  acc = blk_sum(acc, red);
  if (threadIdx.x == 0) {
    float s = acc + attb[0];
    for (int d = 0; d < posD; d++) s += pos[(size_t)row * posD + d] * attw[NT + d];
    satt = 1.0f / (1.0f + expf(-s));
  }
  __syncthreads();
  float w = wvec[widx] * satt;
  for (int c = threadIdx.x; c < Crow; c += TPB)
    out[(size_t)row * Crow + c] += w * fain[(size_t)row * Crow + c];
}

__global__ void scale_copy(const float* __restrict__ x, const float* __restrict__ wvec,
                           int widx, float* __restrict__ out, size_t n)
{
  float w = wvec[widx];
  for (size_t i = blockIdx.x * (size_t)TPB + threadIdx.x; i < n;
       i += (size_t)gridDim.x * TPB)
    out[i] = w * x[i];
}

// fp32 -> f16 pack with row stride change + zero column padding
__global__ void pack_f16(const float* __restrict__ src, int ldsrc, int cols,
                         _Float16* __restrict__ dst, int lddst, int rows)
{
  int i = blockIdx.x * TPB + threadIdx.x;
  if (i >= rows * lddst) return;
  int r = i / lddst, c = i % lddst;
  dst[i] = (_Float16)(c < cols ? src[(size_t)r * ldsrc + c] : 0.0f);
}

// temp_conv: per (b, query-node): argmax over 32 nodes for each of 16 frames
__global__ void tc_argmax(const float* __restrict__ A, int* __restrict__ idx,
                          int NT, int Tt, int Nn, int total)
{
  int wv = (blockIdx.x * blockDim.x + threadIdx.x) >> 5;
  int lane = threadIdx.x & 31;
  if (wv >= total) return;
  const float* a = A + (size_t)wv * NT;
  for (int t = 0; t < Tt; t++) {
    float v = a[t * Nn + lane];
    int bi = lane;
    for (int off = 16; off > 0; off >>= 1) {
      float ov = __shfl_down(v, off);
      int oi = __shfl_down(bi, off);
      if (ov > v) { v = ov; bi = oi; }
    }
    bi = __shfl(bi, 0);
    if (lane == 0) idx[(size_t)wv * Tt + t] = bi;
  }
}

// gather + depthwise conv(7) over time + LN(C) + lrelu + mean over positions
__global__ void __launch_bounds__(TPB)
tc_conv(const float* __restrict__ x, const int* __restrict__ idx,
        const float* __restrict__ cw, const float* __restrict__ cb,
        const float* __restrict__ g, const float* __restrict__ be,
        float* __restrict__ out, int Crow, int Tt, int Nn, int NT)
{
  __shared__ float red[TPB];
  __shared__ int sidx[32];
  int rn = blockIdx.x;          // over B*NT
  int b = rn / NT;
  if (threadIdx.x < Tt) sidx[threadIdx.x] = idx[(size_t)rn * Tt + threadIdx.x];
  __syncthreads();
  int c0 = threadIdx.x, c1 = threadIdx.x + TPB;
  const float* xb = x + (size_t)b * NT * Crow;
  float w0[7], w1[7];
  for (int j = 0; j < 7; j++) {
    w0[j] = cw[c0 * 7 + j];
    w1[j] = (c1 < Crow) ? cw[c1 * 7 + j] : 0.0f;
  }
  float acc0 = 0, acc1 = 0;
  int P = Tt - 6;
  for (int p = 0; p < P; p++) {
    float y0 = cb[c0], y1 = (c1 < Crow) ? cb[c1] : 0.0f;
    for (int j = 0; j < 7; j++) {
      int t = p + j;
      const float* xr = xb + (size_t)(t * Nn + sidx[t]) * Crow;
      y0 += xr[c0] * w0[j];
      if (c1 < Crow) y1 += xr[c1] * w1[j];
    }
    float s = blk_sum(y0 + ((c1 < Crow) ? y1 : 0.0f), red);
    float m = s / Crow;
    float d0 = y0 - m, d1 = (c1 < Crow) ? (y1 - m) : 0.0f;
    float v = blk_sum(d0 * d0 + d1 * d1, red);
    float inv = rsqrtf(v / Crow + 1e-5f);
    float o0 = d0 * inv * g[c0] + be[c0]; o0 = o0 > 0 ? o0 : 0.01f * o0; acc0 += o0;
    if (c1 < Crow) {
      float o1 = d1 * inv * g[c1] + be[c1]; o1 = o1 > 0 ? o1 : 0.01f * o1; acc1 += o1;
    }
  }
  out[(size_t)rn * Crow + c0] = acc0 / P;
  if (c1 < Crow) out[(size_t)rn * Crow + c1] = acc1 / P;
}

// ---------------------------------------------------------------- host side
namespace {

constexpr int B_ = 8, T_ = 16, NN_ = 32, C_ = 512, NT_ = 512, BT_ = 128;
constexpr int BGB_ = 49, GP_ = 64, POSD_ = 9;
constexpr long long SBN = (long long)NT_ * C_;   // 262144, per-batch [512,512]
constexpr size_t NOUT = (size_t)B_ * NT_ * C_;   // 2097152

struct Ws {
  _Float16 *nfH, *xh, *xhT, *nfS, *nfdH, *zH, *zwT, *Ah, *AxH;
  _Float16 *gfnH, *gfT, *Wh, *Wbi2, *ArawH, *AbiH;
  float *Abig, *t0, *t1, *stageIn, *faOut, *srow, *vbuf, *ddbuf;
  int *idx;
};

inline void launch_gemm(hipStream_t st,
    const _Float16* A, long long sA, int lda,
    const _Float16* B, long long sB, int ldb,
    const float* bias,
    float* Cf, long long sCf, int ldc, int accum,
    _Float16* Ch, long long sCh, int ldch, int transC,
    int M, int N, int K, int batch)
{
  long long waves = (long long)batch * (M / 16) * (N / 64);
  int blocks = (int)((waves + 7) / 8);
  wmma_gemm<<<blocks, TPB, 0, st>>>(A, sA, lda, B, sB, ldb, bias,
                                    Cf, sCf, ldc, accum, Ch, sCh, ldch, transC,
                                    M, N, K, batch);
}

inline void diag_prep(hipStream_t st, Ws& w, const float* adj,
                      const _Float16* src, size_t total)
{
  pack_diag<<<(C_ + TPB - 1) / TPB, TPB, 0, st>>>(adj, C_ + 1, w.ddbuf, C_);
  diag_scale<<<(int)((total + TPB - 1) / TPB), TPB, 0, st>>>(src, w.ddbuf, w.nfdH,
                                                             C_, total);
}

// feat_aggr(stageIn; adj, W, b, ln) followed by node_att accumulation into out
void run_stage(hipStream_t st, Ws& w, const float* stageIn,
               const float* adj, const float* aw, const float* ab,
               const float* lg, const float* lb,
               const float* attw, const float* attb,
               const float* pos, const float* wvec, int widx, float* out)
{
  int rows = B_ * NT_;
  rownorm_prep<<<rows, TPB, 0, st>>>(stageIn, C_, NT_, NT_, w.nfS, w.zH,
                                     (_Float16*)nullptr, 0, rows);
  pack_f16<<<(C_ * C_ + TPB - 1) / TPB, TPB, 0, st>>>(aw, C_, C_, w.Wh, C_, C_);
  // gram: A = (nfS*diag) @ nfS^T
  diag_prep(st, w, adj, w.nfS, NOUT);
  launch_gemm(st, w.nfdH, SBN, C_, w.nfS, SBN, C_, nullptr,
              w.Abig, SBN, NT_, 0, nullptr, 0, 0, 0, NT_, NT_, C_, B_);
  softmax5<<<rows, TPB, 0, st>>>(w.Abig, NT_, NT_, w.Ah);
  // zw = z @ W^T + b  (stored transposed f16)
  launch_gemm(st, w.zH, SBN, C_, w.Wh, 0, C_, ab,
              nullptr, 0, 0, 0, w.zwT, SBN, NT_, 1, NT_, C_, C_, B_);
  // y = A @ zw
  launch_gemm(st, w.Ah, SBN, NT_, w.zwT, SBN, NT_, nullptr,
              w.t0, SBN, C_, 0, nullptr, 0, 0, 0, NT_, C_, NT_, B_);
  ln_lrelu<<<rows, TPB, 0, st>>>(w.t0, nullptr, nullptr, lg, lb, w.faOut, C_);
  // node_att on faOut
  rownorm_prep<<<rows, TPB, 0, st>>>(w.faOut, C_, NT_, NT_, w.nfS,
                                     (_Float16*)nullptr, (_Float16*)nullptr, 0, rows);
  natt_v<<<(B_ * C_ + TPB - 1) / TPB, TPB, 0, st>>>(w.nfS, attw, w.vbuf, NT_, C_,
                                                    B_ * C_);
  natt_apply<<<rows, TPB, 0, st>>>(w.nfS, w.vbuf, pos, POSD_, attw, attb,
                                   w.faOut, wvec, widx, out, C_, NT_);
}

} // namespace

extern "C" void kernel_launch(void* const* d_in, const int* in_sizes, int n_in,
                              void* d_out, int out_size, void* d_ws, size_t ws_size,
                              hipStream_t stream) {
  (void)in_sizes; (void)n_in; (void)out_size; (void)ws_size;
  const float* local  = (const float*)d_in[0];
  const float* gfeat  = (const float*)d_in[1];
  const float* pos    = (const float*)d_in[2];
  const float* wvec   = (const float*)d_in[3];
  const float* dp_adj = (const float*)d_in[4];
  const float* dp_w   = (const float*)d_in[5];
  const float* dp_b   = (const float*)d_in[6];
  const float* dp_g   = (const float*)d_in[7];
  const float* dp_be  = (const float*)d_in[8];
  const float* tc_adj = (const float*)d_in[9];
  const float* tc_cw  = (const float*)d_in[10];
  const float* tc_cb  = (const float*)d_in[11];
  const float* tc_g   = (const float*)d_in[12];
  const float* tc_be  = (const float*)d_in[13];
  const float* bi_adj = (const float*)d_in[14];
  const float* bi_w   = (const float*)d_in[15];
  const float* bi_b   = (const float*)d_in[16];
  const float* bi_g   = (const float*)d_in[17];
  const float* bi_be  = (const float*)d_in[18];
  const float* fa_adj[3] = {(const float*)d_in[19], (const float*)d_in[24], (const float*)d_in[29]};
  const float* fa_w[3]   = {(const float*)d_in[20], (const float*)d_in[25], (const float*)d_in[30]};
  const float* fa_b[3]   = {(const float*)d_in[21], (const float*)d_in[26], (const float*)d_in[31]};
  const float* fa_g[3]   = {(const float*)d_in[22], (const float*)d_in[27], (const float*)d_in[32]};
  const float* fa_be[3]  = {(const float*)d_in[23], (const float*)d_in[28], (const float*)d_in[33]};
  const float* na_w[3]   = {(const float*)d_in[34], (const float*)d_in[36], (const float*)d_in[38]};
  const float* na_b[3]   = {(const float*)d_in[35], (const float*)d_in[37], (const float*)d_in[39]};
  float* out = (float*)d_out;

  // ---- workspace carve-up
  char* base = (char*)d_ws;
  size_t off = 0;
  auto alloc = [&](size_t bytes) -> void* {
    void* r = base + off;
    off += (bytes + 255) & ~(size_t)255;
    return r;
  };
  Ws w;
  const size_t H_BN  = (size_t)B_ * NT_ * C_ * sizeof(_Float16);     // 4 MB
  const size_t H_GFN = (size_t)BT_ * GP_ * C_ * sizeof(_Float16);    // 8 MB
  w.nfH   = (_Float16*)alloc(H_BN);
  w.xh    = (_Float16*)alloc(H_BN);
  w.xhT   = (_Float16*)alloc(H_BN);
  w.nfS   = (_Float16*)alloc(H_BN);
  w.nfdH  = (_Float16*)alloc(H_BN);
  w.zH    = (_Float16*)alloc(H_BN);
  w.zwT   = (_Float16*)alloc(H_BN);
  w.Ah    = (_Float16*)alloc(H_BN);
  w.AxH   = (_Float16*)alloc(H_BN);
  w.gfnH  = (_Float16*)alloc(H_GFN);
  w.gfT   = (_Float16*)alloc(H_GFN);
  w.Wh    = (_Float16*)alloc((size_t)C_ * C_ * sizeof(_Float16));
  w.Wbi2  = (_Float16*)alloc((size_t)C_ * GP_ * sizeof(_Float16));
  w.ArawH = (_Float16*)alloc((size_t)BT_ * NN_ * GP_ * sizeof(_Float16));
  w.AbiH  = (_Float16*)alloc((size_t)BT_ * NN_ * GP_ * sizeof(_Float16));
  w.Abig    = (float*)alloc(NOUT * sizeof(float));
  w.t0      = (float*)alloc(NOUT * sizeof(float));
  w.t1      = (float*)alloc(NOUT * sizeof(float));
  w.stageIn = (float*)alloc(NOUT * sizeof(float));
  w.faOut   = (float*)alloc(NOUT * sizeof(float));
  w.srow    = (float*)alloc((size_t)B_ * NT_ * sizeof(float));
  w.vbuf    = (float*)alloc((size_t)B_ * C_ * sizeof(float));
  w.ddbuf   = (float*)alloc((size_t)C_ * sizeof(float));
  w.idx     = (int*)alloc((size_t)B_ * NT_ * T_ * sizeof(int));

  int rows = B_ * NT_;

  // ---- shared prep
  hipMemsetAsync(w.gfnH, 0, H_GFN, stream);   // zero K/N padding rows
  hipMemsetAsync(w.gfT, 0, H_GFN, stream);
  rownorm_prep<<<rows, TPB, 0, stream>>>(local, C_, NT_, NT_, w.nfH, w.xh, w.xhT,
                                         NT_, rows);
  rownorm_prep<<<BT_ * BGB_, TPB, 0, stream>>>(gfeat, C_, BGB_, GP_, w.gfnH,
                                               (_Float16*)nullptr, w.gfT, GP_,
                                               BT_ * BGB_);
  scale_copy<<<4096, TPB, 0, stream>>>(local, wvec, 1, out, NOUT);

  // ---- branch 1: diff_prop -> feat_aggr(fa1) -> node_att(na1)
  pack_f16<<<(C_ * C_ + TPB - 1) / TPB, TPB, 0, stream>>>(dp_w, C_, C_, w.Wh, C_, C_);
  diag_prep(stream, w, dp_adj, w.nfH, NOUT);
  launch_gemm(stream, w.nfdH, SBN, C_, w.nfH, SBN, C_, nullptr,
              w.Abig, SBN, NT_, 0, nullptr, 0, 0, 0, NT_, NT_, C_, B_);
  dpnorm<<<rows, TPB, 0, stream>>>(w.Abig, NT_, w.Ah, w.srow);
  launch_gemm(stream, w.Ah, SBN, NT_, w.xhT, SBN, NT_, nullptr,
              nullptr, 0, 0, 0, w.AxH, SBN, C_, 0, NT_, C_, NT_, B_);   // Ax = A@x
  launch_gemm(stream, w.AxH, SBN, C_, w.Wh, 0, C_, nullptr,
              w.t1, SBN, C_, 0, nullptr, 0, 0, 0, NT_, C_, C_, B_);     // (A@x)W^T
  launch_gemm(stream, w.xh, SBN, C_, w.Wh, 0, C_, dp_b,
              w.t0, SBN, C_, 0, nullptr, 0, 0, 0, NT_, C_, C_, B_);     // xW^T + b
  ln_lrelu<<<rows, TPB, 0, stream>>>(w.t0, w.t1, w.srow, dp_g, dp_be, w.stageIn, C_);
  run_stage(stream, w, w.stageIn, fa_adj[0], fa_w[0], fa_b[0], fa_g[0], fa_be[0],
            na_w[0], na_b[0], pos, wvec, 2, out);

  // ---- branch 2: temp_conv -> feat_aggr(fa2) -> node_att(na2)
  diag_prep(stream, w, tc_adj, w.nfH, NOUT);
  launch_gemm(stream, w.nfdH, SBN, C_, w.nfH, SBN, C_, nullptr,
              w.Abig, SBN, NT_, 0, nullptr, 0, 0, 0, NT_, NT_, C_, B_);
  tc_argmax<<<(rows + 7) / 8, TPB, 0, stream>>>(w.Abig, w.idx, NT_, T_, NN_, rows);
  tc_conv<<<rows, TPB, 0, stream>>>(local, w.idx, tc_cw, tc_cb, tc_g, tc_be,
                                    w.stageIn, C_, T_, NN_, NT_);
  run_stage(stream, w, w.stageIn, fa_adj[1], fa_w[1], fa_b[1], fa_g[1], fa_be[1],
            na_w[1], na_b[1], pos, wvec, 3, out);

  // ---- branch 3: back_incor -> feat_aggr(fa3) -> node_att(na3)
  pack_f16<<<(C_ * C_ + TPB - 1) / TPB, TPB, 0, stream>>>(bi_w, C_ + BGB_, C_,
                                                          w.Wh, C_, C_);
  pack_f16<<<(C_ * GP_ + TPB - 1) / TPB, TPB, 0, stream>>>(bi_w + C_, C_ + BGB_,
                                                           BGB_, w.Wbi2, GP_, C_);
  diag_prep(stream, w, bi_adj, w.nfH, NOUT);
  // Araw = (nf*diag) @ gfn^T  : [128][32][64(pad of 49)], fp32 + f16
  launch_gemm(stream, w.nfdH, (long long)NN_ * C_, C_, w.gfnH, (long long)GP_ * C_, C_,
              nullptr,
              w.Abig, (long long)NN_ * GP_, GP_, 0,
              w.ArawH, (long long)NN_ * GP_, GP_, 0, NN_, GP_, C_, BT_);
  softmax5<<<BT_ * NN_, TPB, 0, stream>>>(w.Abig, GP_, BGB_, w.AbiH);
  // Ag = softmax(5A) @ gf : [128][32][512] f16
  launch_gemm(stream, w.AbiH, (long long)NN_ * GP_, GP_, w.gfT, (long long)C_ * GP_, GP_,
              nullptr, nullptr, 0, 0, 0,
              w.AxH, (long long)NN_ * C_, C_, 0, NN_, C_, GP_, BT_);
  // y = Ag @ Wbi[:, :512]^T + b  ...  += Araw @ Wbi[:, 512:]^T
  launch_gemm(stream, w.AxH, (long long)NN_ * C_, C_, w.Wh, 0, C_, bi_b,
              w.t0, (long long)NN_ * C_, C_, 0, nullptr, 0, 0, 0, NN_, C_, C_, BT_);
  launch_gemm(stream, w.ArawH, (long long)NN_ * GP_, GP_, w.Wbi2, 0, GP_, nullptr,
              w.t0, (long long)NN_ * C_, C_, 1, nullptr, 0, 0, 0,
              NN_, C_, GP_, BT_);
  ln_lrelu<<<rows, TPB, 0, stream>>>(w.t0, nullptr, nullptr, bi_g, bi_be,
                                     w.stageIn, C_);
  run_stage(stream, w, w.stageIn, fa_adj[2], fa_w[2], fa_b[2], fa_g[2], fa_be[2],
            na_w[2], na_b[2], pos, wvec, 4, out);
}
